// GraphAttentionLayer_68607807586524
// MI455X (gfx1250) — matrix operand order repaired and checked by reading, not compiled, not run on hardware
//
#include <hip/hip_runtime.h>

#define N_NODES 8192
#define IN_FEAT 256
#define OUT_FEAT 128
#define ALPHA 0.2f
#define JSTEP 64
#define LDS_STRIDE 72   // bf16 elems per n-row: 64 data + 8 pad -> 144B, uniform 2-way LDS banking

typedef __attribute__((ext_vector_type(16))) __bf16 v16bf;
typedef __attribute__((ext_vector_type(8)))  __bf16 v8bf;
typedef __attribute__((ext_vector_type(8)))  float  v8f;
typedef __attribute__((ext_vector_type(4)))  int    v4i;
typedef __attribute__((ext_vector_type(4)))  float  v4f;

// ---------------------------------------------------------------------------
// Kernel 1: h = x @ trans (fp32 accum). Writes h^T as bf16 [OUT_FEAT][N] into
// workspace (transposed so WMMA B-fragments are contiguous loads), and the
// per-row attention logits e1 = h@a1, e2 = h@a2 (fp32, wave32 shuffle reduce).
// One block of 128 threads handles 16 rows; thread t owns output column t.
// ---------------------------------------------------------------------------
__global__ __launch_bounds__(128) void gat_project(
    const float* __restrict__ x, const float* __restrict__ trans,
    const float* __restrict__ aw, __bf16* __restrict__ hT,
    float* __restrict__ e1, float* __restrict__ e2)
{
    const int t    = threadIdx.x;            // output column 0..127
    const int row0 = blockIdx.x * 16;
    const float a1 = aw[t];
    const float a2 = aw[OUT_FEAT + t];

    float acc[16];
#pragma unroll
    for (int r = 0; r < 16; ++r) acc[r] = 0.f;

    for (int k = 0; k < IN_FEAT; ++k) {
        const float w = trans[k * OUT_FEAT + t];   // coalesced; trans L2-hot
#pragma unroll
        for (int r = 0; r < 16; ++r)               // x[...] uniform -> s_load
            acc[r] = fmaf(x[(size_t)(row0 + r) * IN_FEAT + k], w, acc[r]);
    }

    // transposed bf16 copy for the WMMA B operand
#pragma unroll
    for (int r = 0; r < 16; ++r)
        hT[(size_t)t * N_NODES + row0 + r] = (__bf16)acc[r];

    // e1/e2: reduce across the 128 columns (4 waves)
    __shared__ float r1[4], r2[4];
    const int lane = t & 31, wid = t >> 5;
    for (int r = 0; r < 16; ++r) {
        float v1 = acc[r] * a1;
        float v2 = acc[r] * a2;
#pragma unroll
        for (int off = 16; off > 0; off >>= 1) {
            v1 += __shfl_down(v1, off, 32);
            v2 += __shfl_down(v2, off, 32);
        }
        if (lane == 0) { r1[wid] = v1; r2[wid] = v2; }
        __syncthreads();
        if (t == 0) {
            e1[row0 + r] = r1[0] + r1[1] + r1[2] + r1[3];
            e2[row0 + r] = r2[0] + r2[1] + r2[2] + r2[3];
        }
        __syncthreads();
    }
}

// ---------------------------------------------------------------------------
// Kernel 2: fused masked-softmax attention + P@H, single pass over adj.
// Block = 128 threads = 4 waves; each wave owns a 16-row tile (64 rows/block,
// 128 blocks). Per 64-j chunk: stage h^T[*, jb..jb+63] into LDS (shared by all
// 4 waves), build P in the ISA A-fragment layout from streaming adj (NT loads)
// and e2, then 2 K-steps x 8 n-tiles of v_wmma_f32_16x16x32_bf16.
// rowsum accumulates sum_j p_ij; epilogue divides and applies ELU.
// ---------------------------------------------------------------------------
__global__ __launch_bounds__(128) void gat_attend(
    const int* __restrict__ adj, const __bf16* __restrict__ hT,
    const float* __restrict__ e1, const float* __restrict__ e2,
    float* __restrict__ out)
{
    __shared__ __bf16 lh[OUT_FEAT * LDS_STRIDE];   // 18 KB

    const int t    = threadIdx.x;
    const int lane = t & 31;
    const int wave = t >> 5;
    const int half = (lane >> 4) & 1;   // A: K {0-7,16-23} vs {8-15,24-31}
    const int mrow = lane & 15;         // A/B: M row / N column within tile
    const int row  = blockIdx.x * 64 + wave * 16 + mrow;

    const float e1v = e1[row];
    const int* __restrict__ arow = adj + (size_t)row * N_NODES;

    v8f acc[8];
#pragma unroll
    for (int nt = 0; nt < 8; ++nt)
#pragma unroll
        for (int q = 0; q < 8; ++q) acc[nt][q] = 0.f;
    float rowsum = 0.f;

    for (int jb = 0; jb < N_NODES; jb += JSTEP) {
        // ---- cooperative stage of h^T chunk: thread t copies 128B of col t
        {
            const v4i* src = (const v4i*)(hT + (size_t)t * N_NODES + jb);
            v4i* dst = (v4i*)(&lh[t * LDS_STRIDE]);
#pragma unroll
            for (int q = 0; q < 8; ++q) dst[q] = src[q];
        }
        __syncthreads();

#pragma unroll
        for (int s = 0; s < 2; ++s) {               // two K=32 WMMA steps
            const int j0 = jb + s * 32 + half * 8;  // this lane's first j
            const v4i* pa = (const v4i*)(arow + j0);
            const v4f* pe = (const v4f*)(e2 + j0);
            // adj is a 256MB stream-once: non-temporal, keep L2 for h/e2
            v4i av0 = __builtin_nontemporal_load(pa + 0);
            v4i av1 = __builtin_nontemporal_load(pa + 1);
            v4i av2 = __builtin_nontemporal_load(pa + 4);  // +16 j
            v4i av3 = __builtin_nontemporal_load(pa + 5);
            v4f ev0 = pe[0], ev1 = pe[1], ev2 = pe[4], ev3 = pe[5];

            // p = adj ? exp(leaky_relu(e1_i + e2_j)) : 0   (bounded, no max
            // subtraction needed; masked exp underflows to exactly 0 as in ref)
            auto score = [&](int a, float ev) -> float {
                float sc = e1v + ev;
                sc = sc > 0.f ? sc : ALPHA * sc;
                return a != 0 ? __expf(sc) : 0.f;
            };
            float p[16];
#pragma unroll
            for (int q = 0; q < 4; ++q) {
                p[q]      = score(av0[q], ev0[q]);
                p[4 + q]  = score(av1[q], ev1[q]);
                p[8 + q]  = score(av2[q], ev2[q]);
                p[12 + q] = score(av3[q], ev3[q]);
            }
#pragma unroll
            for (int q = 0; q < 16; ++q) rowsum += p[q];

            v16bf afr;
#pragma unroll
            for (int q = 0; q < 16; ++q) afr[q] = (__bf16)p[q];

#pragma unroll
            for (int nt = 0; nt < 8; ++nt) {
                const int n    = nt * 16 + mrow;        // B column for lane
                const int jloc = s * 32 + half * 8;     // B K {0-7,16-23} vs {8-15,24-31}
                const v8bf* bp = (const v8bf*)(&lh[n * LDS_STRIDE + jloc]);
                v8bf blo = bp[0], bhi = bp[2];
                v16bf bfr;
#pragma unroll
                for (int q = 0; q < 8; ++q) { bfr[q] = blo[q]; bfr[8 + q] = bhi[q]; }
                acc[nt] = __builtin_amdgcn_wmma_f32_16x16x32_bf16(
                    false, afr, false, bfr, (short)0, acc[nt], false, false);
            }
        }
        __syncthreads();
    }

    // combine the two K-halves of each row's sum; lane l now holds the full
    // softmax denominator for row (l & 15)
    rowsum += __shfl_xor(rowsum, 16, 32);

    // epilogue: attn = acc / rowsum, ELU, store fp32
#pragma unroll
    for (int r = 0; r < 8; ++r) {
        const int rl   = r + half * 8;                 // C layout: M for vgpr r
        const float rs = __shfl(rowsum, rl, 32);       // ds_bpermute broadcast
        const int orow = blockIdx.x * 64 + wave * 16 + rl;
#pragma unroll
        for (int nt = 0; nt < 8; ++nt) {
            float a = acc[nt][r] / rs;
            a = a > 0.f ? a : (__expf(a) - 1.f);       // ELU(alpha=1)
            out[(size_t)orow * OUT_FEAT + nt * 16 + mrow] = a;
        }
    }
}

// ---------------------------------------------------------------------------
extern "C" void kernel_launch(void* const* d_in, const int* in_sizes, int n_in,
                              void* d_out, int out_size, void* d_ws, size_t ws_size,
                              hipStream_t stream)
{
    const float* x     = (const float*)d_in[0];   // [8192, 256]
    const int*   adj   = (const int*)d_in[1];     // [8192, 8192]
    const float* trans = (const float*)d_in[2];   // [256, 128]
    const float* aw    = (const float*)d_in[3];   // [256, 1]
    float* out = (float*)d_out;                   // [8192, 128]

    // workspace: h^T bf16 (2MB) | e1 (32KB) | e2 (32KB)
    __bf16* hT = (__bf16*)d_ws;
    float*  e1 = (float*)((char*)d_ws + (size_t)OUT_FEAT * N_NODES * sizeof(__bf16));
    float*  e2 = e1 + N_NODES;

    gat_project<<<N_NODES / 16, 128, 0, stream>>>(x, trans, aw, hT, e1, e2);
    gat_attend <<<N_NODES / 64, 128, 0, stream>>>(adj, hT, e1, e2, out);
}
